// MyLayer_74921409511807
// MI455X (gfx1250) — matrix operand order repaired and checked by reading, not compiled, not run on hardware
//
#include <hip/hip_runtime.h>

#define BATCH 32768
#define UNITS 1024
#define HID   256
#define DT    0.05f
#define STEPS 32

typedef __attribute__((ext_vector_type(16))) __bf16 v16bf;
typedef __attribute__((ext_vector_type(8)))  float  v8f;
typedef __attribute__((ext_vector_type(4)))  unsigned int u32x4;

union FragU { u32x4 q[2]; v16bf v; };

__device__ __forceinline__ unsigned short f2bf(float f) {
    unsigned int u = __float_as_uint(f);
    u += 0x7FFFu + ((u >> 16) & 1u);           // round-to-nearest-even
    return (unsigned short)(u >> 16);
}

__device__ __forceinline__ v8f wmma_bf16(v16bf a, v16bf b, v8f c) {
    // 8 args: (neg_a, A, neg_b, B, c_mod, C, reuse_a, reuse_b)
    return __builtin_amdgcn_wmma_f32_16x16x32_bf16(false, a, false, b, (short)0, c, false, false);
}

// A fragment (16x32 bf16, row-major, K contiguous).  Lane l: m=l%16, half=l/16.
// VGPR v holds K-pair k0=(v&3)*2 + 8*half + 16*(v>>2)  ->  two contiguous 16B chunks.
__device__ __forceinline__ v16bf ldA(const unsigned short* base, int rs, int m, int k0, int half) {
    FragU f;
    f.q[0] = *(const u32x4*)(base + m * rs + k0 + 8 * half);
    f.q[1] = *(const u32x4*)(base + m * rs + k0 + 16 + 8 * half);
    return f.v;
}

// B fragment (32x16 bf16) read from B^T storage [N][K].  Lane l: n=n0+l%16.
// VGPR v holds K-pair k0=(v&3)*2 + 8*half + 16*(v>>2)  ->  two contiguous 16B chunks.
__device__ __forceinline__ v16bf ldB(const unsigned short* base, int rs, int n, int k0, int half) {
    const unsigned short* p = base + (size_t)(n + (threadIdx.x & 15)) * rs + k0;
    FragU f;
    f.q[0] = *(const u32x4*)(p + 8 * half);
    f.q[1] = *(const u32x4*)(p + 16 + 8 * half);
    return f.v;
}

// --- weight prep: bf16 copies in [N][K] (B-transposed) form ------------------
// Bt1 = W1^T [256][1024]   (B of  h  @ W1)
// Bt2 = W2^T [1024][256]   (B of  r  @ W2)
// Bt3 = W2   [256][1024]   (B of  h  @ W2^T)
// Bt4 = W1   [1024][256]   (B of  g  @ W1^T)
__global__ void prep_kernel(const float* __restrict__ W1, const float* __restrict__ W2,
                            unsigned short* __restrict__ Bt1, unsigned short* __restrict__ Bt2,
                            unsigned short* __restrict__ Bt3, unsigned short* __restrict__ Bt4) {
    int idx = blockIdx.x * 256 + threadIdx.x;      // [0, 1024*256)
    Bt4[idx] = f2bf(W1[idx]);
    Bt3[idx] = f2bf(W2[idx]);
    { int n = idx >> 10, k = idx & 1023; Bt1[idx] = f2bf(W1[k * 256 + n]); }
    { int n = idx >> 8,  k = idx & 255;  Bt2[idx] = f2bf(W2[k * 1024 + n]); }
}

// --- fused 32-step RK4 of pvf, 16 rows per workgroup, 8 waves ----------------
__global__ __launch_bounds__(256, 1) void rk4_kernel(
    const float* __restrict__ x, const float* __restrict__ b1, const float* __restrict__ b2,
    const unsigned short* __restrict__ Bt1, const unsigned short* __restrict__ Bt2,
    const unsigned short* __restrict__ Bt3, const unsigned short* __restrict__ Bt4,
    float* __restrict__ out) {

    __shared__ unsigned short heS[16 * UNITS];   // eval point, bf16   (32 KB)
    __shared__ unsigned short rS[16 * HID];      // relu(a)            ( 8 KB)
    __shared__ unsigned short gS[16 * HID];      // mask * (h W2^T)    ( 8 KB)

    const int lane = threadIdx.x & 31;
    const int wave = threadIdx.x >> 5;
    const int l15  = lane & 15;
    const int half = lane >> 4;
    const int mrow = 8 * half;                   // C-frag row offset (add v)
    const int r0   = blockIdx.x * 16;

    // h / acc / k held in registers in the C-fragment layout of a [16,1024]
    // matrix: wave owns cols [128w,128w+128), tile j covers 16 of them.
    float h[64], acc[64];
    v8f kc[8];
    const v8f vzero = {};
#pragma unroll
    for (int j = 0; j < 8; ++j) kc[j] = vzero;

    float b1v[2], b2v[8];
#pragma unroll
    for (int t = 0; t < 2; ++t) b1v[t] = b1[32 * wave + 16 * t + l15];
#pragma unroll
    for (int j = 0; j < 8; ++j) b2v[j] = b2[128 * wave + 16 * j + l15];

#pragma unroll
    for (int j = 0; j < 8; ++j)
#pragma unroll
        for (int v = 0; v < 8; ++v)
            h[j * 8 + v] = x[(size_t)(r0 + v + mrow) * UNITS + 128 * wave + 16 * j + l15];

    const float cvals[4] = {0.0f, 0.5f * DT, 0.5f * DT, DT};
    const float wvals[4] = {DT / 6.0f, DT / 3.0f, DT / 3.0f, DT / 6.0f};

    for (int step = 0; step < STEPS; ++step) {
#pragma unroll
        for (int i = 0; i < 64; ++i) acc[i] = h[i];

#pragma unroll
        for (int e = 0; e < 4; ++e) {
            const float ce = cvals[e], we = wvals[e];

            // eval point he = h + ce * k_prev  -> LDS (bf16)
#pragma unroll
            for (int j = 0; j < 8; ++j)
#pragma unroll
                for (int v = 0; v < 8; ++v) {
                    float val = h[j * 8 + v] + ce * kc[j][v];
                    heS[(v + mrow) * UNITS + 128 * wave + 16 * j + l15] = f2bf(val);
                }
            __syncthreads();

            // GEMM1: a = he @ W1   and   GEMM3: u = he @ W2^T  (N=256, 2 tiles/wave)
            v8f ac[2], uc[2];
#pragma unroll
            for (int t = 0; t < 2; ++t) { ac[t] = vzero; uc[t] = vzero; }
#pragma unroll
            for (int t = 0; t < 2; ++t) {
                const int n0 = 32 * wave + 16 * t;
#pragma unroll 4
                for (int kb = 0; kb < UNITS / 32; ++kb) {
                    v16bf af  = ldA(heS, UNITS, l15, 32 * kb, half);
                    v16bf bw1 = ldB(Bt1, UNITS, n0, 32 * kb, half);
                    ac[t] = wmma_bf16(af, bw1, ac[t]);
                    v16bf bw3 = ldB(Bt3, UNITS, n0, 32 * kb, half);
                    uc[t] = wmma_bf16(af, bw3, uc[t]);
                }
            }

            // r = relu(a+b1),  g = 1[a+b1>0] * u  -> LDS (bf16)
#pragma unroll
            for (int t = 0; t < 2; ++t) {
                const int n = 32 * wave + 16 * t + l15;
#pragma unroll
                for (int v = 0; v < 8; ++v) {
                    float av = ac[t][v] + b1v[t];
                    bool  p  = av > 0.0f;
                    rS[(v + mrow) * HID + n] = f2bf(p ? av : 0.0f);
                    gS[(v + mrow) * HID + n] = f2bf(p ? uc[t][v] : 0.0f);
                }
            }
            __syncthreads();

            // k = r @ W2 + g @ W1^T + b2  (N=1024, 8 tiles/wave, chained C)
#pragma unroll 2
            for (int j = 0; j < 8; ++j) {
                const int n0 = 128 * wave + 16 * j;
                v8f c = vzero;
#pragma unroll 2
                for (int kb = 0; kb < HID / 32; ++kb) {
                    v16bf ar  = ldA(rS, HID, l15, 32 * kb, half);
                    v16bf bw2 = ldB(Bt2, HID, n0, 32 * kb, half);
                    c = wmma_bf16(ar, bw2, c);
                    v16bf ag  = ldA(gS, HID, l15, 32 * kb, half);
                    v16bf bw4 = ldB(Bt4, HID, n0, 32 * kb, half);
                    c = wmma_bf16(ag, bw4, c);
                }
#pragma unroll
                for (int v = 0; v < 8; ++v) c[v] += b2v[j];
                kc[j] = c;
#pragma unroll
                for (int v = 0; v < 8; ++v) acc[j * 8 + v] += we * c[v];
            }
        }

#pragma unroll
        for (int i = 0; i < 64; ++i) h[i] = acc[i];
    }

#pragma unroll
    for (int j = 0; j < 8; ++j)
#pragma unroll
        for (int v = 0; v < 8; ++v)
            out[(size_t)(r0 + v + mrow) * UNITS + 128 * wave + 16 * j + l15] = h[j * 8 + v];
}

extern "C" void kernel_launch(void* const* d_in, const int* in_sizes, int n_in,
                              void* d_out, int out_size, void* d_ws, size_t ws_size,
                              hipStream_t stream) {
    (void)in_sizes; (void)n_in; (void)out_size; (void)ws_size;
    const float* x  = (const float*)d_in[0];
    const float* W1 = (const float*)d_in[1];
    const float* b1 = (const float*)d_in[2];
    const float* W2 = (const float*)d_in[3];
    const float* b2 = (const float*)d_in[4];
    float* out = (float*)d_out;

    unsigned short* base = (unsigned short*)d_ws;   // 4 x 512KB bf16 weight views
    unsigned short* Bt1 = base;
    unsigned short* Bt2 = base + 1 * 256 * 1024;
    unsigned short* Bt3 = base + 2 * 256 * 1024;
    unsigned short* Bt4 = base + 3 * 256 * 1024;

    prep_kernel<<<(UNITS * HID) / 256, 256, 0, stream>>>(W1, W2, Bt1, Bt2, Bt3, Bt4);
    rk4_kernel<<<BATCH / 16, 256, 0, stream>>>(x, b1, b2, Bt1, Bt2, Bt3, Bt4, out);
}
